// DAWN_15668040696249
// MI455X (gfx1250) — matrix operand order repaired and checked by reading, not compiled, not run on hardware
//
#include <hip/hip_runtime.h>
#include <math.h>

// ---------------- WMMA helpers (gfx1250, wave32) ----------------
typedef __attribute__((ext_vector_type(16))) _Float16 v16h;
typedef __attribute__((ext_vector_type(8)))  float    v8f;

#define LDT 34  // LDS row stride (f16 elems) for 32-wide K tiles; even -> dword aligned pairs

union FragU { v16h v; unsigned int u[8]; };

// A fragment: 16x32 f16. lane: M = lane&15, half = lane>>4.
// VGPR i holds K pair at K = (i>>2)*16 + half*8 + (i&3)*2
__device__ __forceinline__ v16h frag_a(const _Float16* lds, int row, int lhalf) {
  FragU f;
  const unsigned short* p = (const unsigned short*)(lds + row * LDT);
#pragma unroll
  for (int i = 0; i < 8; ++i) {
    int k = ((i >> 2) << 4) + (lhalf << 3) + ((i & 3) << 1);
    f.u[i] = *(const unsigned int*)(p + k);
  }
  return f.v;
}

// B fragment: 32x16 f16, staged transposed in LDS as [n][k].
// lane: N = lane&15, half = lane>>4. VGPR j holds K pair at K = (j>>2)*16 + half*8 + (j&3)*2
__device__ __forceinline__ v16h frag_b(const _Float16* lds, int row, int lhalf) {
  FragU f;
  const unsigned short* p = (const unsigned short*)(lds + row * LDT);
#pragma unroll
  for (int i = 0; i < 8; ++i) {
    int k = ((i >> 2) << 4) + (lhalf << 3) + ((i & 3) << 1);
    f.u[i] = *(const unsigned int*)(p + k);
  }
  return f.v;
}

__device__ __forceinline__ v8f v8f_zero() {
  v8f z = {0.f, 0.f, 0.f, 0.f, 0.f, 0.f, 0.f, 0.f};
  return z;
}

// ---------------- GEMM NT: C[M,N] = A[M,K] * W[N,K]^T (+bias,+gelu,+resid) ----------------
// FLAGS: bit0 = bias, bit1 = exact GELU, bit2 = residual add
template<int FLAGS>
__global__ __launch_bounds__(128) void gemm_nt_kernel(
    const float* __restrict__ A, const float* __restrict__ W,
    const float* __restrict__ bias, const float* __restrict__ resid,
    float* __restrict__ C, int M, int N, int K) {
  __shared__ _Float16 As[64 * LDT];
  __shared__ _Float16 Bs[64 * LDT];
  const int t = threadIdx.x;
  const int lane = t & 31, wave = t >> 5;
  const int wm = (wave >> 1) * 32, wn = (wave & 1) * 32;
  const int lrow = lane & 15, lhalf = lane >> 4;
  const int m0 = blockIdx.y * 64, n0 = blockIdx.x * 64;

  v8f acc[2][2];
  acc[0][0] = v8f_zero(); acc[0][1] = v8f_zero();
  acc[1][0] = v8f_zero(); acc[1][1] = v8f_zero();

  for (int k0 = 0; k0 < K; k0 += 32) {
#pragma unroll
    for (int j = 0; j < 16; ++j) {      // 2048 elems each of A,B tile / 128 threads
      int idx = t + j * 128;
      int r = idx >> 5, c = idx & 31;
      As[r * LDT + c] = (_Float16)A[(size_t)(m0 + r) * K + k0 + c];
      Bs[r * LDT + c] = (_Float16)W[(size_t)(n0 + r) * K + k0 + c];
    }
    __syncthreads();
    v16h aF[2], bF[2];
    aF[0] = frag_a(As, wm + lrow, lhalf);
    aF[1] = frag_a(As, wm + 16 + lrow, lhalf);
    bF[0] = frag_b(Bs, wn + lrow, lhalf);
    bF[1] = frag_b(Bs, wn + 16 + lrow, lhalf);
#pragma unroll
    for (int sm = 0; sm < 2; ++sm)
#pragma unroll
      for (int sn = 0; sn < 2; ++sn)
        acc[sm][sn] = __builtin_amdgcn_wmma_f32_16x16x32_f16(
            false, aF[sm], false, bF[sn], (short)0, acc[sm][sn], false, false);
    __syncthreads();
  }
#pragma unroll
  for (int sm = 0; sm < 2; ++sm)
#pragma unroll
    for (int sn = 0; sn < 2; ++sn)
#pragma unroll
      for (int v = 0; v < 8; ++v) {
        int row = m0 + wm + sm * 16 + lhalf * 8 + v;
        int col = n0 + wn + sn * 16 + lrow;
        float val = acc[sm][sn][v];
        if (FLAGS & 1) val += bias[col];
        if (FLAGS & 2) val = 0.5f * val * (1.f + erff(val * 0.70710678118f));
        if (FLAGS & 4) val += resid[(size_t)row * N + col];
        C[(size_t)row * N + col] = val;
      }
}

// ---------------- GEMM NN: C[M,N] = A[M,K] * B[K,N] (B row-major, staged transposed) ----------------
__global__ __launch_bounds__(128) void gemm_nn_kernel(
    const float* __restrict__ A, const float* __restrict__ B,
    float* __restrict__ C, int M, int N, int K) {
  __shared__ _Float16 As[64 * LDT];
  __shared__ _Float16 Bs[64 * LDT];
  const int t = threadIdx.x;
  const int lane = t & 31, wave = t >> 5;
  const int wm = (wave >> 1) * 32, wn = (wave & 1) * 32;
  const int lrow = lane & 15, lhalf = lane >> 4;
  const int m0 = blockIdx.y * 64, n0 = blockIdx.x * 64;

  v8f acc[2][2];
  acc[0][0] = v8f_zero(); acc[0][1] = v8f_zero();
  acc[1][0] = v8f_zero(); acc[1][1] = v8f_zero();

#pragma unroll 1
  for (int k0 = 0; k0 < K; k0 += 32) {
    if (k0 + 32 < K) {  // hint next tiles toward the cache (global_prefetch_b8)
      __builtin_prefetch(&A[(size_t)(m0 + (t >> 1)) * K + k0 + 32 + ((t & 1) << 4)], 0, 1);
      __builtin_prefetch(&B[(size_t)(k0 + 32 + (t >> 2)) * N + n0 + ((t & 3) << 4)], 0, 1);
    }
#pragma unroll
    for (int j = 0; j < 16; ++j) {
      int idx = t + j * 128;
      { int r = idx >> 5, c = idx & 31;                 // A tile: 64 rows x 32 k
        As[r * LDT + c] = (_Float16)A[(size_t)(m0 + r) * K + k0 + c]; }
      { int kk = idx >> 6, nn = idx & 63;               // B tile: 32 k x 64 n, store [n][k]
        Bs[nn * LDT + kk] = (_Float16)B[(size_t)(k0 + kk) * N + n0 + nn]; }
    }
    __syncthreads();
    v16h aF[2], bF[2];
    aF[0] = frag_a(As, wm + lrow, lhalf);
    aF[1] = frag_a(As, wm + 16 + lrow, lhalf);
    bF[0] = frag_b(Bs, wn + lrow, lhalf);
    bF[1] = frag_b(Bs, wn + 16 + lrow, lhalf);
#pragma unroll
    for (int sm = 0; sm < 2; ++sm)
#pragma unroll
      for (int sn = 0; sn < 2; ++sn)
        acc[sm][sn] = __builtin_amdgcn_wmma_f32_16x16x32_f16(
            false, aF[sm], false, bF[sn], (short)0, acc[sm][sn], false, false);
    __syncthreads();
  }
#pragma unroll
  for (int sm = 0; sm < 2; ++sm)
#pragma unroll
    for (int sn = 0; sn < 2; ++sn)
#pragma unroll
      for (int v = 0; v < 8; ++v) {
        int row = m0 + wm + sm * 16 + lhalf * 8 + v;
        int col = n0 + wn + sn * 16 + lrow;
        C[(size_t)row * N + col] = acc[sm][sn][v];
      }
}

// -------- Batched, row-scaled GEMM for d2: sd2[s, n*128+r] = proc_act[s,n] * (x @ down_proj[n])[s,r]
__global__ __launch_bounds__(128) void gemm_d2_kernel(
    const float* __restrict__ A /*x [512,512]*/, const float* __restrict__ Ball /*[128,512,128]*/,
    const float* __restrict__ pact /*[512,128]*/, float* __restrict__ C /*[512,16384]*/,
    int M, int K) {
  __shared__ _Float16 As[64 * LDT];
  __shared__ _Float16 Bs[64 * LDT];
  const int RP = 128, NPROC = 128, LDC = 16384;
  const int t = threadIdx.x;
  const int lane = t & 31, wave = t >> 5;
  const int wm = (wave >> 1) * 32, wn = (wave & 1) * 32;
  const int lrow = lane & 15, lhalf = lane >> 4;
  const int m0 = blockIdx.y * 64, n0 = blockIdx.x * 64;
  const int batch = blockIdx.z;
  const float* B = Ball + (size_t)batch * K * RP;

  v8f acc[2][2];
  acc[0][0] = v8f_zero(); acc[0][1] = v8f_zero();
  acc[1][0] = v8f_zero(); acc[1][1] = v8f_zero();

#pragma unroll 1
  for (int k0 = 0; k0 < K; k0 += 32) {
#pragma unroll
    for (int j = 0; j < 16; ++j) {
      int idx = t + j * 128;
      { int r = idx >> 5, c = idx & 31;
        As[r * LDT + c] = (_Float16)A[(size_t)(m0 + r) * K + k0 + c]; }
      { int kk = idx >> 6, nn = idx & 63;
        Bs[nn * LDT + kk] = (_Float16)B[(size_t)(k0 + kk) * RP + n0 + nn]; }
    }
    __syncthreads();
    v16h aF[2], bF[2];
    aF[0] = frag_a(As, wm + lrow, lhalf);
    aF[1] = frag_a(As, wm + 16 + lrow, lhalf);
    bF[0] = frag_b(Bs, wn + lrow, lhalf);
    bF[1] = frag_b(Bs, wn + 16 + lrow, lhalf);
#pragma unroll
    for (int sm = 0; sm < 2; ++sm)
#pragma unroll
      for (int sn = 0; sn < 2; ++sn)
        acc[sm][sn] = __builtin_amdgcn_wmma_f32_16x16x32_f16(
            false, aF[sm], false, bF[sn], (short)0, acc[sm][sn], false, false);
    __syncthreads();
  }
#pragma unroll
  for (int sm = 0; sm < 2; ++sm)
#pragma unroll
    for (int sn = 0; sn < 2; ++sn)
#pragma unroll
      for (int v = 0; v < 8; ++v) {
        int row = m0 + wm + sm * 16 + lhalf * 8 + v;
        int col = n0 + wn + sn * 16 + lrow;       // within RP
        float val = acc[sm][sn][v] * pact[(size_t)row * NPROC + batch];
        C[(size_t)row * LDC + batch * RP + col] = val;
      }
}

// ---------------- Small kernels ----------------
__global__ __launch_bounds__(256) void embed_kernel(
    const int* __restrict__ ids, const float* __restrict__ tok,
    const float* __restrict__ pos, float* __restrict__ x) {
  int s = blockIdx.x, t = threadIdx.x;
  int id = ids[s];
#pragma unroll
  for (int j = 0; j < 2; ++j) {
    int h = t + j * 256;
    x[(size_t)s * 512 + h] = tok[(size_t)id * 512 + h] + pos[(size_t)s * 512 + h];
  }
}

__global__ __launch_bounds__(128) void rowinv_kernel(
    const float* __restrict__ pat, float* __restrict__ pinv) {
  __shared__ float red[128];
  int n = blockIdx.x, t = threadIdx.x;
  float ss = 0.f;
#pragma unroll
  for (int j = 0; j < 4; ++j) { float v = pat[(size_t)n * 512 + t + j * 128]; ss += v * v; }
  red[t] = ss; __syncthreads();
  for (int o = 64; o > 0; o >>= 1) { if (t < o) red[t] += red[t + o]; __syncthreads(); }
  if (t == 0) pinv[n] = 1.f / fmaxf(sqrtf(red[0]), 1e-12f);
}

__global__ __launch_bounds__(64) void acts_kernel(
    const float* __restrict__ x, const float* __restrict__ pat,
    const float* __restrict__ pinv, float* __restrict__ acts,
    float* __restrict__ actsum) {
  __shared__ float xl[512];
  __shared__ float red[64];
  int s = blockIdx.x, t = threadIdx.x;
  float ss = 0.f;
#pragma unroll
  for (int j = 0; j < 8; ++j) {
    float v = x[(size_t)s * 512 + t + j * 64];
    xl[t + j * 64] = v; ss += v * v;
  }
  red[t] = ss; __syncthreads();
  for (int o = 32; o > 0; o >>= 1) { if (t < o) red[t] += red[t + o]; __syncthreads(); }
  float invx = 1.f / fmaxf(sqrtf(red[0]), 1e-12f);
  __syncthreads();
  float dot = 0.f;
  for (int h = 0; h < 512; ++h) dot += xl[h] * pat[(size_t)t * 512 + h];
  float a = 1.f / (1.f + expf(-dot * invx * pinv[t]));
  acts[(size_t)s * 64 + t] = a;
  red[t] = a; __syncthreads();
  for (int o = 32; o > 0; o >>= 1) { if (t < o) red[t] += red[t + o]; __syncthreads(); }
  if (t == 0) actsum[s] = red[0];
}

// sd[s, n*16+r] = acts[s,n] * sum_h shared[s,h]*ad_down[n,h,r]
__global__ __launch_bounds__(256) void sd_kernel(
    const float* __restrict__ shf, const float* __restrict__ addw,
    const float* __restrict__ acts, float* __restrict__ sd) {
  __shared__ float sh[512];
  int s = blockIdx.x, t = threadIdx.x;
#pragma unroll
  for (int j = 0; j < 2; ++j) sh[t + j * 256] = shf[(size_t)s * 512 + t + j * 256];
  __syncthreads();
#pragma unroll
  for (int j = 0; j < 4; ++j) {
    int o = t + j * 256;
    int n = o >> 4, r = o & 15;
    const float* wp = addw + (size_t)n * 8192 + r;  // [n][h][16]
    float dot = 0.f;
    for (int h = 0; h < 512; ++h) dot += sh[h] * wp[h * 16];
    sd[(size_t)s * 1024 + o] = dot * acts[(size_t)s * 64 + n];
  }
}

// x[s] = LN(x[s] + add1[s] + rowscale[s]*add2[s]) with weight/bias
__global__ __launch_bounds__(256) void add_ln_kernel(
    float* __restrict__ x, const float* __restrict__ add1,
    const float* __restrict__ add2, const float* __restrict__ rowscale,
    const float* __restrict__ w, const float* __restrict__ b) {
  __shared__ float red[256];
  const int H = 512;
  int s = blockIdx.x, t = threadIdx.x;
  float rs = rowscale ? rowscale[s] : 0.f;
  float pre[2]; float lsum = 0.f;
#pragma unroll
  for (int j = 0; j < 2; ++j) {
    int h = t + j * 256;
    float v = x[(size_t)s * H + h];
    if (add1) v += add1[(size_t)s * H + h];
    if (add2) v += add2[(size_t)s * H + h] * rs;
    pre[j] = v; lsum += v;
  }
  red[t] = lsum; __syncthreads();
  for (int o = 128; o > 0; o >>= 1) { if (t < o) red[t] += red[t + o]; __syncthreads(); }
  float mean = red[0] * (1.f / H);
  __syncthreads();
  float lv = 0.f;
#pragma unroll
  for (int j = 0; j < 2; ++j) { float d = pre[j] - mean; lv += d * d; }
  red[t] = lv; __syncthreads();
  for (int o = 128; o > 0; o >>= 1) { if (t < o) red[t] += red[t + o]; __syncthreads(); }
  float inv = rsqrtf(red[0] * (1.f / H) + 1e-5f);
#pragma unroll
  for (int j = 0; j < 2; ++j) {
    int h = t + j * 256;
    x[(size_t)s * H + h] = (pre[j] - mean) * inv * w[h] + b[h];
  }
}

// causal single-head attention over qkv[s, 0:64|64:128|128:192]
__global__ __launch_bounds__(128) void attn_kernel(
    const float* __restrict__ qkv, float* __restrict__ out) {
  __shared__ float qv[64];
  __shared__ float sc[512];
  __shared__ float red[128];
  int q = blockIdx.x, t = threadIdx.x;
  if (t < 64) qv[t] = qkv[(size_t)q * 192 + t];
  __syncthreads();
  int nk = q + 1;
  float lmax = -1e30f;
  for (int k = t; k < nk; k += 128) {
    const float* kp = qkv + (size_t)k * 192 + 64;
    float d = 0.f;
    for (int c = 0; c < 64; ++c) d += qv[c] * kp[c];
    d *= 0.125f;  // 1/sqrt(64)
    sc[k] = d;
    lmax = fmaxf(lmax, d);
  }
  red[t] = lmax; __syncthreads();
  for (int o = 64; o > 0; o >>= 1) { if (t < o) red[t] = fmaxf(red[t], red[t + o]); __syncthreads(); }
  float mx = red[0]; __syncthreads();
  float lsum = 0.f;
  for (int k = t; k < nk; k += 128) { float e = expf(sc[k] - mx); sc[k] = e; lsum += e; }
  red[t] = lsum; __syncthreads();
  for (int o = 64; o > 0; o >>= 1) { if (t < o) red[t] += red[t + o]; __syncthreads(); }
  float inv = 1.f / red[0];
  __syncthreads();
  if (t < 64) {
    float accv = 0.f;
    for (int k = 0; k < nk; ++k) accv += sc[k] * qkv[(size_t)k * 192 + 128 + t];
    out[(size_t)q * 64 + t] = accv * inv;
  }
}

// proc_act[s,p] = sigmoid(conv_b[p] + sum_{tap,c} w[p,tap,c]*enriched[s+tap-2,c])
__global__ __launch_bounds__(128) void conv_kernel(
    const float* __restrict__ enr, const float* __restrict__ w,
    const float* __restrict__ b, float* __restrict__ pact, int S) {
  __shared__ float er[320];
  int s = blockIdx.x, t = threadIdx.x;
  for (int idx = t; idx < 320; idx += 128) {
    int tap = idx >> 6, c = idx & 63;
    int sp = s + tap - 2;
    er[idx] = (sp >= 0 && sp < S) ? enr[(size_t)sp * 64 + c] : 0.f;
  }
  __syncthreads();
  const float* wp = w + (size_t)t * 320;
  float accv = b[t];
  for (int i = 0; i < 320; ++i) accv += wp[i] * er[i];
  pact[(size_t)s * 128 + t] = 1.f / (1.f + expf(-accv));
}

// ---------------- Host orchestration ----------------
extern "C" void kernel_launch(void* const* d_in, const int* in_sizes, int n_in,
                              void* d_out, int out_size, void* d_ws, size_t ws_size,
                              hipStream_t stream) {
  (void)in_sizes; (void)n_in; (void)out_size; (void)ws_size;
  const int S = 512;
  const int* ids        = (const int*)d_in[0];
  const float* tok      = (const float*)d_in[1];
  const float* pos      = (const float*)d_in[2];
  const float* patterns = (const float*)d_in[3];
  const float* st_w     = (const float*)d_in[4];
  const float* st_b     = (const float*)d_in[5];
  const float* ad_down  = (const float*)d_in[6];
  const float* ad_up    = (const float*)d_in[7];
  const float* ln1_w    = (const float*)d_in[8];
  const float* ln1_b    = (const float*)d_in[9];
  const float* in_w     = (const float*)d_in[10];
  const float* in_b     = (const float*)d_in[11];
  const float* ao_w     = (const float*)d_in[12];
  const float* ao_b     = (const float*)d_in[13];
  const float* cv_w     = (const float*)d_in[14];
  const float* cv_b     = (const float*)d_in[15];
  const float* dp       = (const float*)d_in[16];
  const float* up       = (const float*)d_in[17];
  const float* ln2_w    = (const float*)d_in[18];
  const float* ln2_b    = (const float*)d_in[19];
  const float* on_w     = (const float*)d_in[20];
  const float* on_b     = (const float*)d_in[21];
  const float* op_w     = (const float*)d_in[22];
  const float* op_b     = (const float*)d_in[23];

  char* wsp = (char*)d_ws;
  size_t off = 0;
  auto alloc = [&](size_t bytes) -> void* {
    void* p = wsp + off;
    off += (bytes + 255) & ~(size_t)255;
    return p;
  };
  float* x      = (float*)alloc((size_t)S * 512 * 4);
  float* shbuf  = (float*)alloc((size_t)S * 512 * 4);
  float* tmp    = (float*)alloc((size_t)S * 512 * 4);
  float* acts   = (float*)alloc((size_t)S * 64 * 4);
  float* actsum = (float*)alloc((size_t)S * 4);
  float* pinv   = (float*)alloc(64 * 4);
  float* sd     = (float*)alloc((size_t)S * 1024 * 4);
  float* qkv    = (float*)alloc((size_t)S * 192 * 4);
  float* ao     = (float*)alloc((size_t)S * 64 * 4);
  float* enr    = (float*)alloc((size_t)S * 64 * 4);
  float* pact   = (float*)alloc((size_t)S * 128 * 4);
  float* sd2    = (float*)alloc((size_t)S * 16384 * 4);

  embed_kernel<<<S, 256, 0, stream>>>(ids, tok, pos, x);

  for (int l = 0; l < 4; ++l) {
    const float* pat_l = patterns + (size_t)l * 64 * 512;
    // --- BalancedInputNeurons ---
    rowinv_kernel<<<64, 128, 0, stream>>>(pat_l, pinv);
    acts_kernel<<<S, 64, 0, stream>>>(x, pat_l, pinv, acts, actsum);
    gemm_nt_kernel<3><<<dim3(8, 8), 128, 0, stream>>>(     // shared = gelu(x@st_w^T + b)
        x, st_w + (size_t)l * 262144, st_b + (size_t)l * 512, nullptr, shbuf, S, 512, 512);
    sd_kernel<<<S, 256, 0, stream>>>(shbuf, ad_down + (size_t)l * 524288, acts, sd);
    gemm_nn_kernel<<<dim3(8, 8), 128, 0, stream>>>(        // tmp = sd @ ad_up (viewed [1024,512])
        sd, ad_up + (size_t)l * 524288, tmp, S, 512, 1024);
    add_ln_kernel<<<S, 256, 0, stream>>>(x, tmp, shbuf, actsum,
                                         ln1_w + (size_t)l * 512, ln1_b + (size_t)l * 512);
    // --- LateralConnections ---
    gemm_nt_kernel<1><<<dim3(3, 8), 128, 0, stream>>>(     // qkv = acts@in_w^T + b
        acts, in_w + (size_t)l * 12288, in_b + (size_t)l * 192, nullptr, qkv, S, 192, 64);
    attn_kernel<<<S, 128, 0, stream>>>(qkv, ao);
    gemm_nt_kernel<5><<<dim3(1, 8), 128, 0, stream>>>(     // enriched = acts + ao@ao_w^T + b
        ao, ao_w + (size_t)l * 4096, ao_b + (size_t)l * 64, acts, enr, S, 64, 64);
    // --- LowRankProcessNeurons ---
    conv_kernel<<<S, 128, 0, stream>>>(enr, cv_w + (size_t)l * 40960, cv_b + (size_t)l * 128,
                                       pact, S);
    gemm_d2_kernel<<<dim3(2, 8, 128), 128, 0, stream>>>(   // sd2 = pact * (x @ down_proj[n])
        x, dp + (size_t)l * 8388608, pact, sd2, S, 512);
    gemm_nn_kernel<<<dim3(8, 8), 128, 0, stream>>>(        // tmp = sd2 @ up_proj (viewed [16384,512])
        sd2, up + (size_t)l * 8388608, tmp, S, 512, 16384);
    add_ln_kernel<<<S, 256, 0, stream>>>(x, tmp, nullptr, nullptr,
                                         ln2_w + (size_t)l * 512, ln2_b + (size_t)l * 512);
  }

  // final norm + vocab projection
  add_ln_kernel<<<S, 256, 0, stream>>>(x, nullptr, nullptr, nullptr, on_w, on_b);
  gemm_nt_kernel<1><<<dim3(500, 8), 128, 0, stream>>>(
      x, op_w, op_b, nullptr, (float*)d_out, S, 32000, 512);
}
